// TripletLoss_12764642804226
// MI455X (gfx1250) — matrix operand order repaired and checked by reading, not compiled, not run on hardware
//
#include <hip/hip_runtime.h>
#include <stdint.h>

#define NROWS 4096
#define KDIM  2048
#define BM 128
#define BN 128
#define KT 32
#define PAD 8
#define LDT (KT + PAD)     // 40 u16 per LDS row = 80B stride (16B aligned, bank-conflict free)
#define MARGINV 0.3f
#define BIGV 9999999.0f

typedef unsigned short u16;
typedef __attribute__((ext_vector_type(16))) __bf16 v16bf;
typedef __attribute__((ext_vector_type(8)))  float  v8f;

union Frag { uint4 q[2]; v16bf bf; };

// fp32 -> bf16 (round to nearest even)
__device__ __forceinline__ u16 f2bf(float x) {
  unsigned u = __float_as_uint(x);
  unsigned r = u + 0x7FFFu + ((u >> 16) & 1u);
  return (u16)(r >> 16);
}
__device__ __forceinline__ float bf2f(u16 h) {
  return __uint_as_float(((unsigned)h) << 16);
}
// order-preserving float <-> uint keys (for atomic max/min over signed floats)
__device__ __forceinline__ unsigned fkey(float f) {
  unsigned u = __float_as_uint(f);
  return (u & 0x80000000u) ? ~u : (u | 0x80000000u);
}
__device__ __forceinline__ float fval(unsigned k) {
  return __uint_as_float((k & 0x80000000u) ? (k ^ 0x80000000u) : ~k);
}

// Async global -> LDS copy of 16B, tracked by ASYNCcnt (no VGPR round-trip).
// VDST = LDS byte offset (wave-relative; generic shared ptr truncates to it),
// VADDR = 64-bit global address, GV mode.
__device__ __forceinline__ void async_copy_b128(void* lds_dst, const void* gsrc) {
  const unsigned lds_off = (unsigned)(uintptr_t)lds_dst;
  const unsigned long long ga = (unsigned long long)(uintptr_t)gsrc;
  asm volatile("global_load_async_to_lds_b128 %0, %1, off"
               :: "v"(lds_off), "v"(ga)
               : "memory");
}
__device__ __forceinline__ void wait_async0() {
  asm volatile("s_wait_asynccnt 0x0" ::: "memory");
}

// ------- Kernel 1: one-time fp32 -> (hi,lo) bf16 split + row norms + key init -------
__global__ __launch_bounds__(256)
void triplet_split_norms(const float* __restrict__ emb, u16* __restrict__ ghi,
                         u16* __restrict__ glo, float* __restrict__ nrm,
                         unsigned* __restrict__ apk, unsigned* __restrict__ ank) {
  const int row = blockIdx.x;
  const int tid = threadIdx.x;
  const float4* p = (const float4*)(emb + (size_t)row * KDIM);
  float s = 0.f;
#pragma unroll
  for (int i = 0; i < 2; ++i) {
    const int c4 = tid + i * 256;       // float4 index within the row (0..511)
    const float4 x = p[c4];
    s += x.x * x.x + x.y * x.y + x.z * x.z + x.w * x.w;
    const u16 h0 = f2bf(x.x), h1 = f2bf(x.y), h2 = f2bf(x.z), h3 = f2bf(x.w);
    const u16 l0 = f2bf(x.x - bf2f(h0)), l1 = f2bf(x.y - bf2f(h1));
    const u16 l2 = f2bf(x.z - bf2f(h2)), l3 = f2bf(x.w - bf2f(h3));
    const size_t base = (size_t)row * KDIM + (size_t)c4 * 4;
    *(uint2*)(ghi + base) = make_uint2((unsigned)h0 | ((unsigned)h1 << 16),
                                       (unsigned)h2 | ((unsigned)h3 << 16));
    *(uint2*)(glo + base) = make_uint2((unsigned)l0 | ((unsigned)l1 << 16),
                                       (unsigned)l2 | ((unsigned)l3 << 16));
  }
#pragma unroll
  for (int off = 1; off < 32; off <<= 1) s += __shfl_xor(s, off, 32);
  __shared__ float red[8];
  if ((tid & 31) == 0) red[tid >> 5] = s;
  __syncthreads();
  if (tid == 0) {
    float t = 0.f;
#pragma unroll
    for (int j = 0; j < 8; ++j) t += red[j];
    nrm[row] = t;
    apk[row] = 0u;           // below any real key -> atomicMax works
    ank[row] = 0xFFFFFFFFu;  // above any real key -> atomicMin works
  }
}

// ------- Kernel 2: bf16x3 WMMA gram + fused batch-hard mining -------
// Block = 128x128 tile, 8 waves; wave (mw,nw) owns a 32x64 sub-tile:
//   mw = wave>>1 (2 M-tiles of 16 rows), nw = wave&1 (4 N-tiles of 16 cols)
__global__ __launch_bounds__(256)
void triplet_gemm_mine(const u16* __restrict__ ghi, const u16* __restrict__ glo,
                       const int* __restrict__ label, const float* __restrict__ nrm,
                       unsigned* __restrict__ apk, unsigned* __restrict__ ank) {
  __shared__ u16 Ahi[BM][LDT];
  __shared__ u16 Alo[BM][LDT];
  __shared__ u16 Bhi[BN][LDT];
  __shared__ u16 Blo[BN][LDT];
  __shared__ float sRn[BM], sCn[BN];
  __shared__ int   sRl[BM], sCl[BN];

  const int tid = threadIdx.x;
  const int r0 = blockIdx.y * BM;
  const int c0 = blockIdx.x * BN;

  if (tid < BM) {
    sRn[tid] = nrm[r0 + tid];
    sRl[tid] = label[r0 + tid];
  } else {
    const int t2 = tid - BM;
    sCn[t2] = nrm[c0 + t2];
    sCl[t2] = label[c0 + t2];
  }

  const int w    = tid >> 5;
  const int lane = tid & 31;
  const int ln   = lane & 15;
  const int half = lane >> 4;
  const int mw   = w >> 1;  // 0..3
  const int nw   = w & 1;   // 0..1

  v8f acc[8] = {};          // acc[mt*4+t]

  for (int k0 = 0; k0 < KDIM; k0 += KT) {
    __syncthreads();
    // Async staging: 4 planes x 128 rows x 4 x b128 chunks, fire-and-forget,
    // single ASYNCcnt wait before the barrier (no VGPR round-trip, no per-chunk stall).
#pragma unroll
    for (int i = 0; i < 8; ++i) {
      const int arr = i >> 1;                 // 0:Ahi 1:Alo 2:Bhi 3:Blo (compile-time)
      const int rem = (i & 1) * 256 + tid;    // 0..511
      const int row = rem >> 2;               // 0..127
      const int c   = rem & 3;                // 16B chunk (8 u16)
      const size_t gidx =
          (size_t)(((arr >= 2) ? c0 : r0) + row) * KDIM + k0 + c * 8;
      if (arr == 0) {
        async_copy_b128(&Ahi[row][c * 8], ghi + gidx);
      } else if (arr == 1) {
        async_copy_b128(&Alo[row][c * 8], glo + gidx);
      } else if (arr == 2) {
        async_copy_b128(&Bhi[row][c * 8], ghi + gidx);
      } else {
        async_copy_b128(&Blo[row][c * 8], glo + gidx);
      }
    }
    wait_async0();
    __syncthreads();

    // A fragments (16x32 bf16): lane M = ln; elems 0..7 -> K=half*8+e, 8..15 -> K=16+half*8+e
    Frag fa[2][2];
#pragma unroll
    for (int mt = 0; mt < 2; ++mt) {
      const int m  = mw * 32 + mt * 16 + ln;
      const int ka = half * 8;
      fa[mt][0].q[0] = *(const uint4*)&Ahi[m][ka];
      fa[mt][0].q[1] = *(const uint4*)&Ahi[m][ka + 16];
      fa[mt][1].q[0] = *(const uint4*)&Alo[m][ka];
      fa[mt][1].q[1] = *(const uint4*)&Alo[m][ka + 16];
    }
#pragma unroll
    for (int t = 0; t < 4; ++t) {
      // B fragment (32x16 bf16): lane N = ln; elems 0..7 -> K=half*8+e, 8..15 -> K=16+half*8+e
      const int n  = nw * 64 + t * 16 + ln;
      const int kb = half * 8;
      Frag fbh, fbl;
      fbh.q[0] = *(const uint4*)&Bhi[n][kb];
      fbh.q[1] = *(const uint4*)&Bhi[n][kb + 16];
      fbl.q[0] = *(const uint4*)&Blo[n][kb];
      fbl.q[1] = *(const uint4*)&Blo[n][kb + 16];
#pragma unroll
      for (int mt = 0; mt < 2; ++mt) {
        v8f a = acc[mt * 4 + t];
        a = __builtin_amdgcn_wmma_f32_16x16x32_bf16(false, fa[mt][0].bf, false, fbh.bf,
                                                    (short)0, a, false, false);
        a = __builtin_amdgcn_wmma_f32_16x16x32_bf16(false, fa[mt][0].bf, false, fbl.bf,
                                                    (short)0, a, false, false);
        a = __builtin_amdgcn_wmma_f32_16x16x32_bf16(false, fa[mt][1].bf, false, fbh.bf,
                                                    (short)0, a, false, false);
        acc[mt * 4 + t] = a;
      }
    }
  }

  // ---- Epilogue: decode C layout, distances, BIG-masked max/min, atomics ----
  // C 16x16 f32: lanes 0-15 -> N=ln, M=vgpr; lanes 16-31 -> N=ln, M=vgpr+8
#pragma unroll
  for (int mt = 0; mt < 2; ++mt) {
#pragma unroll
    for (int v = 0; v < 8; ++v) {
      const int rl   = mw * 32 + mt * 16 + v + half * 8;
      const float rn = sRn[rl];
      const int   rb = sRl[rl];
      float bap = -3.402823e38f;
      float ban =  3.402823e38f;
#pragma unroll
      for (int t = 0; t < 4; ++t) {
        const int cl  = nw * 64 + t * 16 + ln;
        const float g = acc[mt * 4 + t][v];
        const float d2 = rn + sCn[cl] - 2.0f * g;
        const float d  = sqrtf(fmaxf(d2, 1e-12f));
        const bool sim = (rb == sCl[cl]);
        const float apc = sim ? d : d - BIGV;   // dist - BIG*(1-sim)
        const float anc = sim ? d + BIGV : d;   // dist + BIG*sim
        bap = fmaxf(bap, apc);
        ban = fminf(ban, anc);
      }
#pragma unroll
      for (int off = 1; off < 16; off <<= 1) {
        bap = fmaxf(bap, __shfl_xor(bap, off, 16));
        ban = fminf(ban, __shfl_xor(ban, off, 16));
      }
      if (ln == v) {
        const int row = r0 + rl;
        atomicMax(&apk[row], fkey(bap));
        atomicMin(&ank[row], fkey(ban));
      }
    }
  }
}

// ------- Kernel 3: final scalar reduction -------
__global__ __launch_bounds__(256)
void triplet_finalize(const unsigned* __restrict__ apk, const unsigned* __restrict__ ank,
                      float* __restrict__ out) {
  const int tid = threadIdx.x;
  float ls = 0.f, pc = 0.f;
  for (int r = tid; r < NROWS; r += 256) {
    const float ap = fval(apk[r]);
    const float an = fval(ank[r]);
    const float l = MARGINV - (an - ap);
    ls += (l > 0.f) ? l : 0.f;
    pc += (an > ap) ? 1.f : 0.f;
  }
#pragma unroll
  for (int off = 1; off < 32; off <<= 1) {
    ls += __shfl_xor(ls, off, 32);
    pc += __shfl_xor(pc, off, 32);
  }
  __shared__ float rl_[8], rp_[8];
  if ((tid & 31) == 0) { rl_[tid >> 5] = ls; rp_[tid >> 5] = pc; }
  __syncthreads();
  if (tid == 0) {
    float L = 0.f, P = 0.f;
#pragma unroll
    for (int j = 0; j < 8; ++j) { L += rl_[j]; P += rp_[j]; }
    out[0] = L / (float)NROWS;
    out[1] = P / (float)NROWS;
  }
}

extern "C" void kernel_launch(void* const* d_in, const int* in_sizes, int n_in,
                              void* d_out, int out_size, void* d_ws, size_t ws_size,
                              hipStream_t stream) {
  const float* emb   = (const float*)d_in[0];
  const int*   label = (const int*)d_in[1];
  float* out = (float*)d_out;

  // workspace: ghi[4096*2048] u16 | glo[4096*2048] u16 | nrm f32[4096] | apk u32[4096] | ank u32[4096]
  u16* ghi = (u16*)d_ws;
  u16* glo = ghi + (size_t)NROWS * KDIM;
  float* nrm = (float*)(glo + (size_t)NROWS * KDIM);
  unsigned* apk = (unsigned*)(nrm + NROWS);
  unsigned* ank = apk + NROWS;

  triplet_split_norms<<<NROWS, 256, 0, stream>>>(emb, ghi, glo, nrm, apk, ank);
  triplet_gemm_mine<<<dim3(NROWS / BN, NROWS / BM), 256, 0, stream>>>(ghi, glo, label, nrm,
                                                                      apk, ank);
  triplet_finalize<<<1, 256, 0, stream>>>(apk, ank, out);
}
